// MultiHeadAttention_66769561584162
// MI455X (gfx1250) — matrix operand order repaired and checked by reading, not compiled, not run on hardware
//
#include <hip/hip_runtime.h>

typedef _Float16 f16;
typedef __attribute__((ext_vector_type(16))) _Float16 v16h;
typedef __attribute__((ext_vector_type(8)))  _Float16 v8h;
typedef __attribute__((ext_vector_type(4)))  _Float16 v4h;
typedef __attribute__((ext_vector_type(8)))  float    v8f;
typedef __attribute__((ext_vector_type(4)))  float    v4f;
typedef unsigned int u32x4 __attribute__((ext_vector_type(4)));
typedef int          i32x4 __attribute__((ext_vector_type(4)));
typedef int          i32x8 __attribute__((ext_vector_type(8)));

#define TDIM 4096
#define SDIM 4096
#define BDIM 4
#define EDIM 512

#if __has_builtin(__builtin_amdgcn_tensor_load_to_lds)
#define HAS_TDM 1
#else
#define HAS_TDM 0
#endif

__device__ __forceinline__ v8f wmma_f32_f16(v16h a, v16h b, v8f c) {
  // D = A(16x32 f16) * B(32x16 f16) + C(16x16 f32)
  return __builtin_amdgcn_wmma_f32_16x16x32_f16(false, a, false, b, (short)0, c,
                                                false, false);
}

__device__ __forceinline__ v16h mkfrag(v8h lo, v8h hi) {
  v16h r;
#pragma unroll
  for (int i = 0; i < 8; ++i) { r[i] = lo[i]; r[i + 8] = hi[i]; }
  return r;
}

// Async 16B copy global -> LDS (GLOBAL_LOAD_ASYNC_TO_LDS_B128, ASYNCcnt).
// Low 32 bits of a generic LDS-aperture pointer are the LDS byte address
// (ISA 10.2).
__device__ __forceinline__ void async_ld_b128(void* lds_dst, const void* gsrc) {
  asm volatile("global_load_async_to_lds_b128 %0, %1, off"
               :
               : "v"((unsigned int)(uintptr_t)lds_dst),
                 "v"((unsigned long long)(uintptr_t)gsrc)
               : "memory");
}

__device__ __forceinline__ void wait_async0() {
#if __has_builtin(__builtin_amdgcn_s_wait_asynccnt)
  __builtin_amdgcn_s_wait_asynccnt(0);
#else
  asm volatile("s_wait_asynccnt 0x0" ::: "memory");
#endif
}

__device__ __forceinline__ void wait_tensor0() {
#if __has_builtin(__builtin_amdgcn_s_wait_tensorcnt)
  __builtin_amdgcn_s_wait_tensorcnt(0);
#else
  asm volatile("s_wait_tensorcnt 0x0" ::: "memory");
#endif
}

#if HAS_TDM
// TDM: 1-D f16 tile (n elems, n <= 65535) global -> LDS, inserting 4 DWORDs of
// LDS padding after every 256 DWORDs (i.e. stride 520 halfs per 512-half row).
__device__ __forceinline__ void tdm_load_1d_f16(f16* lds_dst, const f16* gsrc,
                                                unsigned n) {
  unsigned long long ga = (unsigned long long)(uintptr_t)gsrc;
  u32x4 g0 = {};
  i32x8 g1 = {};
  i32x4 gz = {};
  g0[0] = 1u;                                  // count=1, user descriptor
  g0[1] = (unsigned int)(uintptr_t)lds_dst;    // lds_addr
  g0[2] = (unsigned int)ga;                    // global_addr[31:0]
  g0[3] = (unsigned int)((ga >> 32) & 0x01FFFFFFu) | 0x80000000u;  // type=2
  // [17:16] data_size=1(2B), [20] pad_en, [24:22] pad_interval=7 (256 DW),
  // [31:25] pad_amount=3 (4 DW)
  g1[0] = (int)0x07D10000u;
  g1[1] = (int)((n & 0xFFFFu) << 16);          // tensor_dim0 lo16 (<<16)
  g1[2] = (int)((n >> 16) | (1u << 16));       // tensor_dim0 hi16 | tensor_dim1=1
  g1[3] = (int)((n & 0xFFFFu) << 16);          // tile_dim0 = n
  g1[4] = 0;                                   // tile_dim1/2 unused
  g1[5] = (int)n;                              // tensor_dim0_stride
  g1[6] = 0;
  g1[7] = 0;
#if __clang_major__ >= 23
  i32x8 g4 = {};
  __builtin_amdgcn_tensor_load_to_lds(g0, g1, gz, gz, g4, 0);
#else
  __builtin_amdgcn_tensor_load_to_lds(g0, g1, gz, gz, 0);
#endif
}
#endif

// ---------------------------------------------------------------------------
// fp32 -> f16 conversion (vectorized x4)
// ---------------------------------------------------------------------------
__global__ __launch_bounds__(256) void cvt_kernel(const float* __restrict__ in,
                                                  f16* __restrict__ out, int n4) {
  int i = blockIdx.x * 256 + threadIdx.x;
  if (i < n4) {
    v4f v = *(const v4f*)(in + (size_t)i * 4);
    v4h h;
#pragma unroll
    for (int j = 0; j < 4; ++j) h[j] = (f16)v[j];
    *(v4h*)(out + (size_t)i * 4) = h;
  }
}

// ---------------------------------------------------------------------------
// GEMM: out = A(M x 512, f16 row-major) * Wt^T + bias
//   Wt is [N=512][K=512] row-major (torch weight W[f][e] directly).
//   mode 0: out16[((i&3)*TDIM + (i>>2))*512 + n]   (b-major Q/K)
//   mode 1: out16[((i&3)*512 + n)*SDIM + (i>>2)]   (b-major V transposed)
//   mode 2: out32[i*512 + n]                       (final fp32 output)
// Block tile 64(M) x 128(N); 8 waves as 4(M) x 2(N-halves of 64).
// Double-buffered LDS filled by async-to-LDS copies: 1 barrier per K-step.
// ---------------------------------------------------------------------------
__global__ __launch_bounds__(256) void proj_gemm_kernel(
    const f16* __restrict__ A, const f16* __restrict__ Wt,
    const float* __restrict__ bias, f16* __restrict__ out16,
    float* __restrict__ out32, int mode) {
  constexpr int K = EDIM;
  constexpr int LDT = 40;  // 32 + 8 halfs padding
  __shared__ __align__(16) f16 sA[2][64 * LDT];
  __shared__ __align__(16) f16 sB[2][128 * LDT];

  const int tid = threadIdx.x;
  const int lane = tid & 31;
  const int wave = tid >> 5;
  const int M0 = blockIdx.x * 64;
  const int N0 = blockIdx.y * 128;
  const int lm = lane & 15, lh = lane >> 4;
  const int m_base = (wave & 3) * 16;
  const int n_base = (wave >> 2) * 64;

  auto stageAB = [&](int buf, int kk) {
    {  // A tile 64 x 32
      int row = tid >> 2, seg = tid & 3;
      async_ld_b128(&sA[buf][row * LDT + seg * 8],
                    A + (size_t)(M0 + row) * K + kk + seg * 8);
    }
    {  // B tile (K-major) 128 x 32
      int row = tid >> 1, seg = tid & 1;
      const f16* src = Wt + (size_t)(N0 + row) * K + kk + seg * 16;
      f16* dst = &sB[buf][row * LDT + seg * 16];
      async_ld_b128(dst, src);
      async_ld_b128(dst + 8, src + 8);
    }
  };

  v8f acc[4];
  v8f vz = {};
#pragma unroll
  for (int i = 0; i < 4; ++i) acc[i] = vz;

  stageAB(0, 0);
  wait_async0();
  __syncthreads();

  for (int kk = 0; kk < K; kk += 32) {
    const int cur = (kk >> 5) & 1;
    if (kk + 32 < K) stageAB(cur ^ 1, kk + 32);  // DMA overlaps WMMAs below

    const f16* ap = sA[cur] + (m_base + lm) * LDT;
    v16h af = mkfrag(*(const v8h*)(ap + 16 * lh), *(const v8h*)(ap + 16 * lh + 8));
#pragma unroll
    for (int nt = 0; nt < 4; ++nt) {
      const f16* bp = sB[cur] + (n_base + nt * 16 + lm) * LDT + 16 * lh;
      v16h bf = mkfrag(*(const v8h*)(bp), *(const v8h*)(bp + 8));
      acc[nt] = wmma_f32_f16(af, bf, acc[nt]);
    }
    wait_async0();
    __syncthreads();
  }

#pragma unroll
  for (int nt = 0; nt < 4; ++nt) {
    const int n = N0 + n_base + nt * 16 + lm;
    const float bv = bias[n];
#pragma unroll
    for (int r = 0; r < 8; ++r) {
      const int i = M0 + m_base + r + 8 * lh;  // flat input row = t*B + b
      const float val = acc[nt][r] + bv;
      if (mode == 2) {
        out32[(size_t)i * EDIM + n] = val;
      } else {
        const int t = i >> 2, b = i & 3;  // BDIM == 4
        if (mode == 0)
          out16[((size_t)b * TDIM + t) * EDIM + n] = (f16)val;
        else
          out16[((size_t)b * EDIM + n) * SDIM + t] = (f16)val;
      }
    }
  }
}

// ---------------------------------------------------------------------------
// Pass 1: per-(b,s) column stats of logits = (QK^T + mask) * inv_sqrt_d
// (softmax is over the T axis: reference's softmax(dim=1) quirk).
// Block: (b, 128 s-columns). 8 waves x 16 columns; each wave streams all T.
// Q/mask tiles double-buffered in LDS via async copies; K rows direct from
// global (K is 4 MB/batch -> resident in the 192 MB L2).
// ---------------------------------------------------------------------------
__global__ __launch_bounds__(256) void attn_stats_kernel(
    const f16* __restrict__ Qh, const f16* __restrict__ Kh,
    const float* __restrict__ mask, float* __restrict__ mout,
    float* __restrict__ zout) {
  constexpr int LDQ = 520;
  __shared__ __align__(16) f16 sQ[2][16 * LDQ];
  __shared__ __align__(16) float sMask[2][16 * 128];

  const int b = blockIdx.y;
  const int S0 = blockIdx.x * 128;
  const int tid = threadIdx.x, lane = tid & 31, wave = tid >> 5;
  const int lm = lane & 15, lh = lane >> 4;
  const int s_local = wave * 16 + lm;
  const int s = S0 + s_local;
  const float scale = 0.04419417382415922f;  // 1/sqrt(512)

  const f16* krow = Kh + ((size_t)b * SDIM + s) * EDIM;  // this lane's column

  auto stage = [&](int buf, int t0) {
    for (int idx = tid; idx < 16 * 64; idx += 256) {
      int row = idx >> 6, seg = idx & 63;
      async_ld_b128(&sQ[buf][row * LDQ + seg * 8],
                    Qh + ((size_t)b * TDIM + t0 + row) * EDIM + seg * 8);
    }
    for (int idx = tid; idx < 16 * 32; idx += 256) {
      int row = idx >> 5, c4 = idx & 31;
      async_ld_b128(&sMask[buf][row * 128 + c4 * 4],
                    mask + (size_t)(t0 + row) * SDIM + S0 + c4 * 4);
    }
  };

  float run_m = -3.0e38f, run_z = 0.f;

  stage(0, 0);
  wait_async0();
  __syncthreads();

  for (int t0 = 0; t0 < TDIM; t0 += 16) {
    const int cur = (t0 >> 4) & 1;
    if (t0 + 16 < TDIM) stage(cur ^ 1, t0 + 16);  // prefetch next tile

    v8f acc = {};
#pragma unroll
    for (int kk = 0; kk < EDIM; kk += 32) {
      const f16* ap = sQ[cur] + lm * LDQ + kk;
      v16h af = mkfrag(*(const v8h*)(ap + 16 * lh), *(const v8h*)(ap + 16 * lh + 8));
      const f16* bp = krow + kk + 16 * lh;
      v16h bf = mkfrag(*(const v8h*)(bp), *(const v8h*)(bp + 8));
      acc = wmma_f32_f16(af, bf, acc);
    }

    float lg[8], tile_m = -3.0e38f;
#pragma unroll
    for (int r = 0; r < 8; ++r) {
      lg[r] = (acc[r] + sMask[cur][(r + 8 * lh) * 128 + s_local]) * scale;
      tile_m = fmaxf(tile_m, lg[r]);
    }
    const float nm = fmaxf(run_m, tile_m);
    float za = 0.f;
#pragma unroll
    for (int r = 0; r < 8; ++r) za += __expf(lg[r] - nm);
    run_z = run_z * __expf(run_m - nm) + za;
    run_m = nm;

    wait_async0();
    __syncthreads();
  }

  // combine the two lane-halves holding the same column (rows r vs r+8)
  const float om = __shfl_xor(run_m, 16, 32);
  const float oz = __shfl_xor(run_z, 16, 32);
  const float fm = fmaxf(run_m, om);
  const float fz = run_z * __expf(run_m - fm) + oz * __expf(om - fm);
  if (lh == 0) {
    mout[(size_t)b * SDIM + s] = fm;
    zout[(size_t)b * SDIM + s] = fz;
  }
}

// ---------------------------------------------------------------------------
// Pass 2: attn[b,t,:] = sum_s exp((q.k+mask)*scale - m[s]) / z[s] * V[b,s,:]
// Block: (b, 32 t-rows). Stream S in 64-wide tiles.
//   Phase 1: 8 waves compute the 2x4 grid of 16x16 P tiles -> exp -> f16 LDS.
//   Phase 2: 8 waves as 2(t16) x 4(E128); accumulate full E=512 in registers.
// Q block staged once via the Tensor Data Mover (with TDM LDS padding to get
// the conflict-free 520-half stride); mask tiles via async-to-LDS copies that
// overlap the phase-1 WMMAs.
// ---------------------------------------------------------------------------
__global__ __launch_bounds__(256) void attn_av_kernel(
    const f16* __restrict__ Qh, const f16* __restrict__ Kh,
    const f16* __restrict__ Vt, const float* __restrict__ mask,
    const float* __restrict__ mstat, const float* __restrict__ zstat,
    f16* __restrict__ attn_out) {
  constexpr int LDQ = 520;
  constexpr int LDP = 72;
  __shared__ __align__(16) f16 sQ[32 * LDQ];
  __shared__ __align__(16) f16 sP[32 * LDP];
  __shared__ __align__(16) float sMask[32 * 64];

  const int b = blockIdx.y;
  const int T0 = blockIdx.x * 32;
  const int tid = threadIdx.x, lane = tid & 31, wave = tid >> 5;
  const int lm = lane & 15, lh = lane >> 4;
  const float scale = 0.04419417382415922f;

#if HAS_TDM
  if (wave == 0) {
    tdm_load_1d_f16(sQ, Qh + ((size_t)b * TDIM + T0) * EDIM, 32u * EDIM);
    wait_tensor0();
  }
#else
  for (int idx = tid; idx < 32 * 64; idx += 256) {
    int row = idx >> 6, seg = idx & 63;
    async_ld_b128(&sQ[row * LDQ + seg * 8],
                  Qh + ((size_t)b * TDIM + T0 + row) * EDIM + seg * 8);
  }
  wait_async0();
#endif
  __syncthreads();

  const int tt = wave >> 2, ss = wave & 3;  // phase-1 tile coords (2x4)
  const int tw = wave >> 2, ew = wave & 3;  // phase-2 slot (t16, E128)

  v8f acc[8];
  v8f vz = {};
#pragma unroll
  for (int i = 0; i < 8; ++i) acc[i] = vz;

  for (int s0 = 0; s0 < SDIM; s0 += 64) {
    // async mask tile 32x64 (overlaps the phase-1 WMMAs below)
    for (int idx = tid; idx < 32 * 16; idx += 256) {
      int row = idx >> 4, c4 = idx & 15;
      async_ld_b128(&sMask[row * 64 + c4 * 4],
                    mask + (size_t)(T0 + row) * SDIM + s0 + c4 * 4);
    }

    // phase 1: P tile = Q(16x512) * K^T(512x16)
    v8f pc = {};
#pragma unroll
    for (int kk = 0; kk < EDIM; kk += 32) {
      const f16* ap = sQ + (tt * 16 + lm) * LDQ + kk;
      v16h af = mkfrag(*(const v8h*)(ap + 16 * lh), *(const v8h*)(ap + 16 * lh + 8));
      const f16* bp =
          Kh + ((size_t)b * SDIM + s0 + ss * 16 + lm) * EDIM + kk + 16 * lh;
      v16h bf = mkfrag(*(const v8h*)(bp), *(const v8h*)(bp + 8));
      pc = wmma_f32_f16(af, bf, pc);
    }
    wait_async0();
    __syncthreads();  // sMask ready; previous phase-2 done with sP

    {
      const int sl = ss * 16 + lm;
      const int s = s0 + sl;
      const float mm = mstat[(size_t)b * SDIM + s];
      const float iz = 1.f / zstat[(size_t)b * SDIM + s];
#pragma unroll
      for (int r = 0; r < 8; ++r) {
        const int tl = tt * 16 + r + 8 * lh;
        const float lgt = (pc[r] + sMask[tl * 64 + sl]) * scale;
        sP[tl * LDP + sl] = (f16)(__expf(lgt - mm) * iz);
      }
    }
    __syncthreads();  // sP ready

    // phase 2: acc(16x128) += P(16x64) * V(64x128)  [V pre-transposed [e][s]]
#pragma unroll
    for (int kc = 0; kc < 2; ++kc) {
      const f16* ap = sP + (tw * 16 + lm) * LDP + kc * 32;
      v16h af = mkfrag(*(const v8h*)(ap + 16 * lh), *(const v8h*)(ap + 16 * lh + 8));
#pragma unroll
      for (int nt = 0; nt < 8; ++nt) {
        const int e = ew * 128 + nt * 16 + lm;
        const f16* bp =
            Vt + ((size_t)b * EDIM + e) * SDIM + s0 + kc * 32 + 16 * lh;
        v16h bf = mkfrag(*(const v8h*)(bp), *(const v8h*)(bp + 8));
        acc[nt] = wmma_f32_f16(af, bf, acc[nt]);
      }
    }
    __syncthreads();  // before next iter overwrites sP / sMask
  }

#pragma unroll
  for (int nt = 0; nt < 8; ++nt) {
    const int e = ew * 128 + nt * 16 + lm;
#pragma unroll
    for (int r = 0; r < 8; ++r) {
      const int t = T0 + tw * 16 + r + 8 * lh;
      attn_out[((size_t)t * BDIM + b) * EDIM + e] = (f16)acc[nt][r];
    }
  }
}

// ---------------------------------------------------------------------------
extern "C" void kernel_launch(void* const* d_in, const int* in_sizes, int n_in,
                              void* d_out, int out_size, void* d_ws,
                              size_t ws_size, hipStream_t stream) {
  (void)in_sizes; (void)n_in; (void)out_size; (void)ws_size;
  const float* query = (const float*)d_in[0];
  const float* key   = (const float*)d_in[1];
  const float* value = (const float*)d_in[2];
  const float* mask  = (const float*)d_in[3];
  const float* Wq = (const float*)d_in[4];
  const float* bq = (const float*)d_in[5];
  const float* Wk = (const float*)d_in[6];
  const float* bk = (const float*)d_in[7];
  const float* Wv = (const float*)d_in[8];
  const float* bv = (const float*)d_in[9];
  const float* Wo = (const float*)d_in[10];
  const float* bo = (const float*)d_in[11];
  float* out = (float*)d_out;

  const size_t M = (size_t)TDIM * BDIM;             // 16384 rows
  const size_t XSZ = M * EDIM * sizeof(f16);        // 16 MB
  const size_t WSZ = (size_t)EDIM * EDIM * sizeof(f16);

  char* p = (char*)d_ws;
  auto take = [&](size_t n) {
    char* r = p;
    p += (n + 255) & ~(size_t)255;
    return r;
  };
  f16* qraw = (f16*)take(XSZ);
  f16* kraw = (f16*)take(XSZ);
  f16* vraw = (f16*)take(XSZ);
  f16* Wq16 = (f16*)take(WSZ);
  f16* Wk16 = (f16*)take(WSZ);
  f16* Wv16 = (f16*)take(WSZ);
  f16* Wo16 = (f16*)take(WSZ);
  f16* Qh = (f16*)take(XSZ);   // [b][t][e]
  f16* Kh = (f16*)take(XSZ);   // [b][s][e]
  f16* Vt = (f16*)take(XSZ);   // [b][e][s]
  float* mbuf = (float*)take((size_t)BDIM * SDIM * sizeof(float));
  float* zbuf = (float*)take((size_t)BDIM * SDIM * sizeof(float));
  f16* attn16 = qraw;  // qraw is dead after the Q projection; reuse as attn

  const int n4x = (int)(M * EDIM / 4);
  cvt_kernel<<<n4x / 256, 256, 0, stream>>>(query, qraw, n4x);
  cvt_kernel<<<n4x / 256, 256, 0, stream>>>(key, kraw, n4x);
  cvt_kernel<<<n4x / 256, 256, 0, stream>>>(value, vraw, n4x);
  const int n4w = EDIM * EDIM / 4;
  cvt_kernel<<<n4w / 256, 256, 0, stream>>>(Wq, Wq16, n4w);
  cvt_kernel<<<n4w / 256, 256, 0, stream>>>(Wk, Wk16, n4w);
  cvt_kernel<<<n4w / 256, 256, 0, stream>>>(Wv, Wv16, n4w);
  cvt_kernel<<<n4w / 256, 256, 0, stream>>>(Wo, Wo16, n4w);

  dim3 gp((unsigned)(M / 64), (unsigned)(EDIM / 128));
  proj_gemm_kernel<<<gp, 256, 0, stream>>>(qraw, Wq16, bq, Qh, nullptr, 0);
  proj_gemm_kernel<<<gp, 256, 0, stream>>>(kraw, Wk16, bk, Kh, nullptr, 0);
  proj_gemm_kernel<<<gp, 256, 0, stream>>>(vraw, Wv16, bv, Vt, nullptr, 1);

  attn_stats_kernel<<<dim3(SDIM / 128, BDIM), 256, 0, stream>>>(Qh, Kh, mask,
                                                                mbuf, zbuf);
  attn_av_kernel<<<dim3(TDIM / 32, BDIM), 256, 0, stream>>>(Qh, Kh, Vt, mask,
                                                            mbuf, zbuf, attn16);

  proj_gemm_kernel<<<gp, 256, 0, stream>>>(attn16, Wo16, bo, nullptr, out, 2);
}
